// CausalSelfAttention_4226247819300
// MI455X (gfx1250) — compile-verified, matrix operand order checked by reading, not run on hardware
//
#include <hip/hip_runtime.h>
#include <hip/hip_bf16.h>

typedef __attribute__((ext_vector_type(16))) _Float16 v16h;
typedef __attribute__((ext_vector_type(8)))  _Float16 v8h;
typedef __attribute__((ext_vector_type(2)))  _Float16 v2h;
typedef __attribute__((ext_vector_type(8)))  float    v8f;

// Problem dims (compile-time so index math is shifts/masks, not idiv)
constexpr int Bn = 4, Tt = 2048, Cc = 1024, Hh = 16, Dd = 64;
constexpr int Mdim = Bn * Tt;          // 8192
constexpr int Ndim = Cc, Kdim = Cc;    // 1024, 1024

__device__ __forceinline__ v8f zero8() {
  v8f z;
#pragma unroll
  for (int i = 0; i < 8; ++i) z[i] = 0.0f;
  return z;
}

// CDNA5 async copy: global -> LDS, 16 bytes per lane, tracked by ASYNCcnt.
// LDS byte offset = low 32 bits of the generic pointer (ISA: LDS aperture
// address truncates to addr[31:0]).
__device__ __forceinline__ void async_ld_b128(_Float16* lds, const _Float16* g) {
  unsigned loff = (unsigned)(size_t)lds;
  asm volatile("global_load_async_to_lds_b128 %0, %1, off"
               :: "v"(loff), "v"(g) : "memory");
}
__device__ __forceinline__ void wait_async() {
  asm volatile("s_wait_asynccnt 0x0" ::: "memory");
}

// 16x32 f16 A/B WMMA fragment from a K-contiguous row (ld halfs between rows).
// CDNA5 layout: lanes 0-15 hold K 0..7 & 16..23; lanes 16-31 hold K 8..15 & 24..31.
__device__ __forceinline__ v16h load_frag(const _Float16* __restrict__ base,
                                          int row, int ld, int kb, int lane) {
  const int g = (lane >> 4) & 1;
  const _Float16* p = base + (size_t)row * ld + kb + g * 8;
  v8h lo = *(const v8h*)(p);
  v8h hi = *(const v8h*)(p + 16);
  v16h r;
#pragma unroll
  for (int i = 0; i < 8; ++i) { r[i] = lo[i]; r[8 + i] = hi[i]; }
  return r;
}

// Vectorized f32 -> f16 conversion: n must be a multiple of 8.
__global__ void cvt_f32_to_f16_k(const float* __restrict__ in,
                                 _Float16* __restrict__ out, long n8) {
  long i = (long)blockIdx.x * blockDim.x + threadIdx.x;
  const long stride = (long)gridDim.x * blockDim.x;
  for (; i < n8; i += stride) {
    const float4 a = ((const float4*)in)[i * 2 + 0];
    const float4 b = ((const float4*)in)[i * 2 + 1];
    v8h o;
    o[0] = (_Float16)a.x; o[1] = (_Float16)a.y; o[2] = (_Float16)a.z; o[3] = (_Float16)a.w;
    o[4] = (_Float16)b.x; o[5] = (_Float16)b.y; o[6] = (_Float16)b.z; o[7] = (_Float16)b.w;
    *(v8h*)(out + i * 8) = o;
  }
}

// C[M,N] = scale * (A[M,K] @ W[N,K]^T), double-buffered async-LDS-staged WMMA GEMM.
// OUT_MODE 0: f32 row-major [M,N].  OUT_MODE 1: f16 [B,H,T,D] (m=b*T+t, n=h*D+dd).
template <int OUT_MODE>
__global__ __launch_bounds__(256) void gemm_xwT_wmma(
    const _Float16* __restrict__ A, const _Float16* __restrict__ W,
    void* __restrict__ out, float scale) {
  constexpr int BM = 128, BN = 128, BK = 32;
  __shared__ _Float16 sA[2][BM * BK];  // 8 KB each
  __shared__ _Float16 sB[2][BN * BK];

  const int tid  = threadIdx.x;
  const int lane = tid & 31;
  const int wave = tid >> 5;
  const int wm = wave & 1;   // 2 waves along M (64 rows each)
  const int wn = wave >> 1;  // 4 waves along N (32 cols each)
  const int mBase = blockIdx.x * BM;
  const int nBase = blockIdx.y * BN;

  // Stage one K-slab (BMxBK of A, BNxBK of W) into LDS buffer `buf` (async DMA).
  auto stage = [&](int buf, int kb) {
#pragma unroll
    for (int c = 0; c < 2; ++c) {
      const int idx = tid * 16 + c * 8;       // 0..4095 step 8
      const int r = idx >> 5, col = idx & 31; // row 0..127, col 0..31
      async_ld_b128(&sA[buf][idx], &A[(size_t)(mBase + r) * Kdim + kb + col]);
      async_ld_b128(&sB[buf][idx], &W[(size_t)(nBase + r) * Kdim + kb + col]);
      if (kb + 2 * BK < Kdim) {  // prefetch 2 slabs ahead -> global_prefetch_b8
        __builtin_prefetch(&A[(size_t)(mBase + r) * Kdim + kb + 2 * BK + col], 0, 1);
        __builtin_prefetch(&W[(size_t)(nBase + r) * Kdim + kb + 2 * BK + col], 0, 1);
      }
    }
  };

  v8f acc[4][2];
#pragma unroll
  for (int ms = 0; ms < 4; ++ms)
#pragma unroll
    for (int ns = 0; ns < 2; ++ns) acc[ms][ns] = zero8();

  stage(0, 0);
  int buf = 0;
  for (int kb = 0; kb < Kdim; kb += BK) {
    wait_async();     // this wave's staged slab is in LDS
    __syncthreads();  // all waves' slabs in; prior reads of buf^1 done
    if (kb + BK < Kdim) stage(buf ^ 1, kb + BK);  // overlaps with compute below

    v16h af[4], bf[2];
#pragma unroll
    for (int ms = 0; ms < 4; ++ms)
      af[ms] = load_frag(&sA[buf][0], wm * 64 + ms * 16 + (lane & 15), BK, 0, lane);
#pragma unroll
    for (int ns = 0; ns < 2; ++ns)
      bf[ns] = load_frag(&sB[buf][0], wn * 32 + ns * 16 + (lane & 15), BK, 0, lane);
#pragma unroll
    for (int ms = 0; ms < 4; ++ms)
#pragma unroll
      for (int ns = 0; ns < 2; ++ns)
        acc[ms][ns] = __builtin_amdgcn_wmma_f32_16x16x32_f16(
            false, af[ms], false, bf[ns], (short)0, acc[ms][ns], false, false);
    buf ^= 1;
  }

  const int moff = (lane >> 4) ? 8 : 0;
  const int m0 = mBase + wm * 64;
  const int n0 = nBase + wn * 32;
#pragma unroll
  for (int ms = 0; ms < 4; ++ms) {
#pragma unroll
    for (int ns = 0; ns < 2; ++ns) {
#pragma unroll
      for (int i = 0; i < 8; ++i) {
        const int m = m0 + ms * 16 + i + moff;
        const int n = n0 + ns * 16 + (lane & 15);
        const float v = acc[ms][ns][i] * scale;
        if (OUT_MODE == 0) {
          ((float*)out)[(size_t)m * Ndim + n] = v;
        } else {
          const int b = m >> 11, t = m & (Tt - 1);   // T = 2048
          const int h = n >> 6,  dd = n & (Dd - 1);  // D = 64
          ((_Float16*)out)[(((size_t)b * Hh + h) * Tt + t) * Dd + dd] = (_Float16)v;
        }
      }
    }
  }
}

// Flash attention: grid (B*H, T/64), 128 threads (4 waves); wave owns 16 q rows.
// Q pre-scaled by log2(e)/sqrt(D): softmax runs in the exp2 domain so every
// exponential is a bare v_exp_f32. Writes y as f16 [B*T, C].
__global__ __launch_bounds__(128) void attn_wmma(
    const _Float16* __restrict__ Q, const _Float16* __restrict__ Km,
    const _Float16* __restrict__ V, _Float16* __restrict__ Y) {
  __shared__ _Float16 ldsK[32 * 64];     // K tile   [kv=32][d=64]
  __shared__ _Float16 ldsVt[64 * 32];    // V^T tile [d=64][kv=32]
  __shared__ _Float16 ldsP[4][16 * 32];  // per-wave P tile [16][32]

  const int lane = threadIdx.x & 31;
  const int wave = threadIdx.x >> 5;
  const int bh = blockIdx.x;
  const int q0 = blockIdx.y * 64;
  const int qw = q0 + wave * 16;
  const int b = bh >> 4, h = bh & (Hh - 1);

  const _Float16* Qb = Q + (size_t)bh * Tt * Dd;
  const _Float16* Kb = Km + (size_t)bh * Tt * Dd;
  const _Float16* Vb = V + (size_t)bh * Tt * Dd;

  v16h qf[2];
  qf[0] = load_frag(Qb, qw + (lane & 15), Dd, 0, lane);
  qf[1] = load_frag(Qb, qw + (lane & 15), Dd, 32, lane);

  v16h ones;  // all-ones B fragment: row-sum accumulator column (l = P @ 1)
#pragma unroll
  for (int e = 0; e < 16; ++e) ones[e] = (_Float16)1.0f;

  float mrow[8];
#pragma unroll
  for (int i = 0; i < 8; ++i) mrow[i] = -3.0e38f;
  // oacc[0..3] = O tiles (d subtiles), oacc[4] = softmax denominator (P @ ones)
  v8f oacc[5];
#pragma unroll
  for (int d = 0; d < 5; ++d) oacc[d] = zero8();

  const int moff = (lane >> 4) ? 8 : 0;

  // One kv tile (32 keys). `masked` is a call-site constant: the bulk phase
  // (kv0 + 31 < q0 <= every row in the block) carries zero masking code.
  auto kv_step = [&](int kv0, bool masked) {
    // Stage K tile via async global->LDS DMA; V transposed through registers.
#pragma unroll
    for (int c = 0; c < 2; ++c) {
      const int idx = (threadIdx.x * 2 + c) * 8;  // 0..2047 step 8
      const int r = idx >> 6, col = idx & 63;
      async_ld_b128(&ldsK[idx], &Kb[(size_t)(kv0 + r) * Dd + col]);
      v8h vv = *(const v8h*)&Vb[(size_t)(kv0 + r) * Dd + col];
#pragma unroll
      for (int e = 0; e < 8; ++e) ldsVt[(col + e) * 32 + r] = vv[e];
    }
    wait_async();
    __syncthreads();

    // S = Q @ K^T : two 16x16 N-subtiles, K-dim 64.
    v8f s[2];
    s[0] = zero8(); s[1] = zero8();
#pragma unroll
    for (int ns = 0; ns < 2; ++ns)
#pragma unroll
      for (int kc = 0; kc < 2; ++kc) {
        v16h kf = load_frag(ldsK, ns * 16 + (lane & 15), 64, kc * 32, lane);
        s[ns] = __builtin_amdgcn_wmma_f32_16x16x32_f16(
            false, qf[kc], false, kf, (short)0, s[ns], false, false);
      }

    // (Optional) causal mask; per-row tile max candidates.
    float mx[8];
#pragma unroll
    for (int i = 0; i < 8; ++i) {
      float s0 = s[0][i], s1 = s[1][i];
      if (masked) {
        const int qrow = qw + i + moff;
        const int kcol = kv0 + (lane & 15);
        s0 = (kcol > qrow) ? -3.0e38f : s0;
        s1 = (kcol + 16 > qrow) ? -3.0e38f : s1;
        s[0][i] = s0; s[1][i] = s1;
      }
      mx[i] = fmaxf(s0, s1);
    }
    // 16-lane max reduction, rows packed in pairs as 2xf16 (v_pk_max_num_f16).
#pragma unroll
    for (int j = 0; j < 4; ++j) {
      v2h pk;
      pk[0] = (_Float16)mx[2 * j];
      pk[1] = (_Float16)mx[2 * j + 1];
      float f = __builtin_bit_cast(float, pk);
#pragma unroll
      for (int off = 8; off >= 1; off >>= 1) {
        const float o = __shfl_xor(f, off, 32);
        v2h a = __builtin_bit_cast(v2h, f);
        v2h c = __builtin_bit_cast(v2h, o);
        a = __builtin_elementwise_max(a, c);
        f = __builtin_bit_cast(float, a);
      }
      const v2h r = __builtin_bit_cast(v2h, f);
      mx[2 * j] = (float)r[0];
      mx[2 * j + 1] = (float)r[1];
    }

    // Online-softmax update in the exp2 domain; stage P for layout change.
#pragma unroll
    for (int i = 0; i < 8; ++i) {
      const float mnew = fmaxf(mrow[i], mx[i]);
      const float corr = __builtin_exp2f(mrow[i] - mnew);
      mrow[i] = mnew;
      const float p0 = __builtin_exp2f(s[0][i] - mnew);
      const float p1 = __builtin_exp2f(s[1][i] - mnew);
#pragma unroll
      for (int d = 0; d < 5; ++d) oacc[d][i] *= corr;
      ldsP[wave][(i + moff) * 32 + (lane & 15)] = (_Float16)p0;
      ldsP[wave][(i + moff) * 32 + 16 + (lane & 15)] = (_Float16)p1;
    }
    asm volatile("" ::: "memory");  // keep P stores before reloads (wave-local LDS)

    // O += P @ V ; denominator += P @ ones
    v16h pf = load_frag(&ldsP[wave][0], lane & 15, 32, 0, lane);
#pragma unroll
    for (int d = 0; d < 4; ++d) {
      v16h vf = load_frag(ldsVt, d * 16 + (lane & 15), 32, 0, lane);
      oacc[d] = __builtin_amdgcn_wmma_f32_16x16x32_f16(
          false, pf, false, vf, (short)0, oacc[d], false, false);
    }
    oacc[4] = __builtin_amdgcn_wmma_f32_16x16x32_f16(
        false, pf, false, ones, (short)0, oacc[4], false, false);
    __syncthreads();
  };

  // Bulk phase: kv tiles entirely below the q tile -> no causal masking.
  for (int kv0 = 0; kv0 < q0; kv0 += 32) kv_step(kv0, false);
  // Boundary phase: the two tiles overlapping the diagonal.
  for (int kv0 = q0; kv0 < q0 + 64; kv0 += 32) kv_step(kv0, true);

  // Finalize and write y[B*T, C] (f16).
#pragma unroll
  for (int i = 0; i < 8; ++i) {
    const float inv = __builtin_amdgcn_rcpf(oacc[4][i]);
    const int qrow = qw + i + moff;
#pragma unroll
    for (int d = 0; d < 4; ++d) {
      const int dd = d * 16 + (lane & 15);
      Y[((size_t)b * Tt + qrow) * Cc + h * Dd + dd] = (_Float16)(oacc[d][i] * inv);
    }
  }
}

extern "C" void kernel_launch(void* const* d_in, const int* in_sizes, int n_in,
                              void* d_out, int out_size, void* d_ws, size_t ws_size,
                              hipStream_t stream) {
  const float* x  = (const float*)d_in[0];
  const float* Wq = (const float*)d_in[1];
  const float* Wk = (const float*)d_in[2];
  const float* Wv = (const float*)d_in[3];
  const float* Wp = (const float*)d_in[4];

  char* w = (char*)d_ws;
  size_t off = 0;
  auto alloc = [&](size_t bytes) -> void* {
    void* p = w + off;
    off += (bytes + 255) & ~(size_t)255;
    return p;
  };
  _Float16* xh  = (_Float16*)alloc((size_t)Mdim * Cc * 2);
  _Float16* wqh = (_Float16*)alloc((size_t)Cc * Cc * 2);
  _Float16* wkh = (_Float16*)alloc((size_t)Cc * Cc * 2);
  _Float16* wvh = (_Float16*)alloc((size_t)Cc * Cc * 2);
  _Float16* wph = (_Float16*)alloc((size_t)Cc * Cc * 2);
  _Float16* Qh  = (_Float16*)alloc((size_t)Mdim * Cc * 2);
  _Float16* Kh  = (_Float16*)alloc((size_t)Mdim * Cc * 2);
  _Float16* Vh  = (_Float16*)alloc((size_t)Mdim * Cc * 2);
  _Float16* Yh  = (_Float16*)alloc((size_t)Mdim * Cc * 2);

  // 1) f32 -> f16 conversions (vectorized, n/8 work-items)
  cvt_f32_to_f16_k<<<2048, 256, 0, stream>>>(x, xh, (long)Mdim * Cc / 8);
  cvt_f32_to_f16_k<<<512, 256, 0, stream>>>(Wq, wqh, (long)Cc * Cc / 8);
  cvt_f32_to_f16_k<<<512, 256, 0, stream>>>(Wk, wkh, (long)Cc * Cc / 8);
  cvt_f32_to_f16_k<<<512, 256, 0, stream>>>(Wv, wvh, (long)Cc * Cc / 8);
  cvt_f32_to_f16_k<<<512, 256, 0, stream>>>(Wp, wph, (long)Cc * Cc / 8);

  // 2) Q/K/V projections -> [B,H,T,D] f16.
  //    Q is pre-scaled by log2(e)/sqrt(D) so attention exponentials are exp2.
  const dim3 ggrid(Mdim / 128, Ndim / 128);
  gemm_xwT_wmma<1><<<ggrid, 256, 0, stream>>>(xh, wqh, (void*)Qh, 0.125f * 1.44269504f);
  gemm_xwT_wmma<1><<<ggrid, 256, 0, stream>>>(xh, wkh, (void*)Kh, 1.0f);
  gemm_xwT_wmma<1><<<ggrid, 256, 0, stream>>>(xh, wvh, (void*)Vh, 1.0f);

  // 3) causal flash attention -> Yh [B*T, C] f16
  attn_wmma<<<dim3(Bn * Hh, Tt / 64), 128, 0, stream>>>(Qh, Kh, Vh, Yh);

  // 4) output projection -> d_out f32
  gemm_xwT_wmma<0><<<ggrid, 256, 0, stream>>>(Yh, wph, d_out, 1.0f);
}
